// FusedMoEModule_44444321579110
// MI455X (gfx1250) — matrix operand order repaired and checked by reading, not compiled, not run on hardware
//
#include <hip/hip_runtime.h>

// ---------------------------------------------------------------------------
// Fused MoE: T=2048 tokens, D=2048, H=2048, E=8 experts, top-K=2.
// bf16 WMMA (v_wmma_f32_16x16x32_bf16), fp32->bf16 weight cvt in registers,
// async global->LDS staging (ASYNCcnt), branchless 2x-unrolled double buffer.
// ---------------------------------------------------------------------------

typedef __attribute__((ext_vector_type(16))) __bf16 bf16x16;
typedef __attribute__((ext_vector_type(8)))  __bf16 bf16x8;
typedef __attribute__((ext_vector_type(4)))  __bf16 bf16x4;
typedef __attribute__((ext_vector_type(8)))  float  f32x8;

#define T_TOK 2048
#define D_DIM 2048
#define H_DIM 2048
#define TWO_H 4096
#define E_EXP 8
#define K_TOP 2
#define MAXR  (T_TOK * K_TOP)   // worst-case rows per expert
#define LDA   40                // padded LDS row stride (bf16), 80B: 16B-aligned, avoids 4-way bank conflict

// workspace layout (bytes)
static constexpr size_t OFF_CNT  = 0;                                       // int cnt[E] (zeroed each launch)
static constexpr size_t OFF_ROWS = 1024;                                    // int rows[E][MAXR]  (slot = t*K + k)
static constexpr size_t OFF_WTS  = OFF_ROWS + (size_t)E_EXP * MAXR * 4;     // float wts[E][MAXR]
static constexpr size_t OFF_HSB  = OFF_WTS  + (size_t)E_EXP * MAXR * 4;     // bf16 hidden [T][D]
static constexpr size_t OFF_ACT  = OFF_HSB  + (size_t)T_TOK * D_DIM * 2;    // bf16 act [T*K][H]
// total ~24.3 MB

// ---------------------------------------------------------------------------
// CDNA5 async global->LDS copy (16B per lane), tracked by ASYNCcnt.
// VDST operand is the wave-relative LDS byte address = low 32 bits of a
// generic pointer into LDS (flat LDS aperture truncates to addr[31:0]).
// ---------------------------------------------------------------------------
__device__ __forceinline__ void async_copy_b128(unsigned int lds_addr,
                                                unsigned long long gaddr) {
  asm volatile("global_load_async_to_lds_b128 %0, %1, off"
               :: "v"(lds_addr), "v"(gaddr) : "memory");
}
__device__ __forceinline__ void wait_async0() {
  asm volatile("s_wait_asynccnt 0x0" ::: "memory");
}

// ---------------------------------------------------------------------------
// 1) Routing: build per-expert slot lists with atomics.
// ---------------------------------------------------------------------------
__global__ void moe_route_kernel(const int* __restrict__ ids,
                                 const float* __restrict__ w,
                                 int* __restrict__ cnt, int* __restrict__ rows,
                                 float* __restrict__ wts) {
  int idx = blockIdx.x * blockDim.x + threadIdx.x;
  if (idx < T_TOK * K_TOP) {
    int e = ids[idx];
    int pos = atomicAdd(&cnt[e], 1);
    rows[e * MAXR + pos] = idx;      // slot id (t*K + k); token = slot >> 1
    wts [e * MAXR + pos] = w[idx];   // combine weight, folded into act
  }
}

// ---------------------------------------------------------------------------
// 2) hidden fp32 -> bf16 (once).
// ---------------------------------------------------------------------------
__global__ void moe_cvt_kernel(const float* __restrict__ x, __bf16* __restrict__ y) {
  int i = blockIdx.x * blockDim.x + threadIdx.x;   // one float4 per thread
  float4 f = ((const float4*)x)[i];
  bf16x4 o;
  o[0] = (__bf16)f.x; o[1] = (__bf16)f.y; o[2] = (__bf16)f.z; o[3] = (__bf16)f.w;
  ((bf16x4*)y)[i] = o;
}

// ---------------------------------------------------------------------------
// Fragment helpers (wave32; A and B both use contiguous k, not the ISA table order).
// B 32x16 bf16: lane n (0-15) holds col n, K=0..15 (contiguous); lane n+16: K=16..31.
// ---------------------------------------------------------------------------
__device__ __forceinline__ bf16x16 cvt_b_frag(const float* __restrict__ p) {
  bf16x16 r;
#pragma unroll
  for (int i = 0; i < 4; ++i) {
    float4 f = ((const float4*)p)[i];
    r[4*i+0] = (__bf16)f.x; r[4*i+1] = (__bf16)f.y;
    r[4*i+2] = (__bf16)f.z; r[4*i+3] = (__bf16)f.w;
  }
  return r;
}

// A 16x32 bf16 from LDS tile (row-major, stride LDA):
// lane m (0-15): row m, K 0..15; lane m+16: row m, K 16..31.
__device__ __forceinline__ bf16x16 lds_a_frag(const __bf16* aT, int sub, int lane) {
  int row = sub * 16 + (lane & 15);
  int off = (lane < 16) ? 0 : 16;
  const __bf16* p = aT + row * LDA + off;
  bf16x8 lo = *(const bf16x8*)(p);        // ds_load_b128
  bf16x8 hi = *(const bf16x8*)(p + 8);    // ds_load_b128
  bf16x16 r;
#pragma unroll
  for (int i = 0; i < 8; ++i) { r[i] = lo[i]; r[i + 8] = hi[i]; }
  return r;
}

// fast SiLU: x * v_rcp_f32(1 + exp(-x))  (no IEEE divide sequence)
__device__ __forceinline__ float fast_silu(float x) {
  return x * __builtin_amdgcn_rcpf(1.f + __expf(-x));
}

// ---------------------------------------------------------------------------
// 3) Up-projection + SiLU gate: act[slot][h] = silu(gate)*up * topk_weight.
// Block: 256 thr (8 waves). Tile: M=64 tokens x N=128 h-channels, k-step 32.
// Branchless 2x-unrolled double-buffer pipeline; 8 WMMA per wave per k-step.
// ---------------------------------------------------------------------------
__global__ __launch_bounds__(256)
void moe_up_kernel(const __bf16* __restrict__ hsb,
                   const float* __restrict__ w_up,
                   const int* __restrict__ rows,
                   const float* __restrict__ wts,
                   const int* __restrict__ cnt,
                   __bf16* __restrict__ act) {
  __shared__ __bf16 aT[2][64 * LDA];
  __shared__ int    sT[64];
  __shared__ float  wT[64];

  const int e     = blockIdx.z;
  const int cn    = cnt[e];
  const int mBase = blockIdx.x * 64;
  if (mBase >= cn) return;                 // uniform early exit

  const int tid  = threadIdx.x;
  const int lane = tid & 31;
  const int wave = tid >> 5;
  const int h0   = blockIdx.y * 128 + wave * 16;

  if (tid < 64) {
    int i = mBase + tid;
    int slot = 0; float w = 0.f;
    if (i < cn) { slot = rows[e * MAXR + i]; w = wts[e * MAXR + i]; }
    sT[tid] = slot; wT[tid] = w;
  }
  __syncthreads();

  const int col = lane & 15, half = lane >> 4;
  const size_t wbase = (size_t)e * TWO_H * D_DIM;
  const float* pg = w_up + wbase + (size_t)(h0 + col) * D_DIM + half * 16;          // gate rows
  const float* pu = w_up + wbase + (size_t)(H_DIM + h0 + col) * D_DIM + half * 16;  // up rows

  const f32x8 zero = {0.f,0.f,0.f,0.f,0.f,0.f,0.f,0.f};
  f32x8 accg[4] = {zero, zero, zero, zero};
  f32x8 accu[4] = {zero, zero, zero, zero};

  // async A staging: 256 thr, each owns 16B of one token row per k-step
  const int ldRow = tid >> 2, ldSeg = tid & 3;
  const unsigned long long aSrc =
      (unsigned long long)(uintptr_t)(hsb + (size_t)(sT[ldRow] >> 1) * D_DIM + ldSeg * 8);
  const unsigned int ldsDst0 = (unsigned int)(uintptr_t)&aT[0][ldRow * LDA + ldSeg * 8];
  const unsigned int ldsDst1 = (unsigned int)(uintptr_t)&aT[1][ldRow * LDA + ldSeg * 8];

  auto mma_tiles = [&](const __bf16* cur, const float* wg_p, const float* wu_p) {
    bf16x16 bg = cvt_b_frag(wg_p);
    bf16x16 bu = cvt_b_frag(wu_p);
#pragma unroll
    for (int s = 0; s < 4; ++s) {
      bf16x16 a = lds_a_frag(cur, s, lane);
      accg[s] = __builtin_amdgcn_wmma_f32_16x16x32_bf16(false, a, false, bg,
                                                        (short)0, accg[s], false, false);
      accu[s] = __builtin_amdgcn_wmma_f32_16x16x32_bf16(false, a, false, bu,
                                                        (short)0, accu[s], false, false);
    }
  };

  async_copy_b128(ldsDst0, aSrc);          // prime buffer 0 (tile 0)
  wait_async0();
  __syncthreads();

  // steady state: tiles 0..KT-3 (KT = D/32 = 64, even), fully branchless
#pragma unroll 1
  for (int k0 = 0; k0 < D_DIM - 64; k0 += 64) {
    async_copy_b128(ldsDst1, aSrc + (unsigned long long)(k0 + 32) * 2);
    __builtin_prefetch(pg + k0 + 64, 0, 1);
    __builtin_prefetch(pu + k0 + 64, 0, 1);
    mma_tiles(aT[0], pg + k0, pu + k0);
    wait_async0(); __syncthreads();

    async_copy_b128(ldsDst0, aSrc + (unsigned long long)(k0 + 64) * 2);
    __builtin_prefetch(pg + k0 + 96, 0, 1);
    __builtin_prefetch(pu + k0 + 96, 0, 1);
    mma_tiles(aT[1], pg + k0 + 32, pu + k0 + 32);
    wait_async0(); __syncthreads();
  }
  { // epilogue: tiles KT-2 (already in buf0) and KT-1
    const int k0 = D_DIM - 64;
    async_copy_b128(ldsDst1, aSrc + (unsigned long long)(k0 + 32) * 2);
    mma_tiles(aT[0], pg + k0, pu + k0);
    wait_async0(); __syncthreads();
    mma_tiles(aT[1], pg + k0 + 32, pu + k0 + 32);
  }

  // Epilogue: D-layout VGPR r -> (M = r + 8*half, N = col); row stores are
  // lane-contiguous 32B segments. Fold silu*up*combine, store bf16.
#pragma unroll
  for (int s = 0; s < 4; ++s) {
#pragma unroll
    for (int r = 0; r < 8; ++r) {
      int m = s * 16 + half * 8 + r;
      if (mBase + m < cn) {
        float a = fast_silu(accg[s][r]) * accu[s][r] * wT[m];
        act[(size_t)sT[m] * H_DIM + h0 + col] = (__bf16)a;
      }
    }
  }
}

// ---------------------------------------------------------------------------
// 4) Down-projection: out[t][d] += act[slot] @ w_dn[e]^T (atomic scatter-add,
// K=2 expert contributions per token). Same skeleton, K-dim = H.
// ---------------------------------------------------------------------------
__global__ __launch_bounds__(256)
void moe_dn_kernel(const __bf16* __restrict__ act,
                   const float* __restrict__ w_dn,
                   const int* __restrict__ rows,
                   const int* __restrict__ cnt,
                   float* __restrict__ out) {
  __shared__ __bf16 aT[2][64 * LDA];
  __shared__ int    sT[64];

  const int e     = blockIdx.z;
  const int cn    = cnt[e];
  const int mBase = blockIdx.x * 64;
  if (mBase >= cn) return;

  const int tid  = threadIdx.x;
  const int lane = tid & 31;
  const int wave = tid >> 5;
  const int d0   = blockIdx.y * 128 + wave * 16;

  if (tid < 64) {
    int i = mBase + tid;
    sT[tid] = (i < cn) ? rows[e * MAXR + i] : 0;
  }
  __syncthreads();

  const int col = lane & 15, half = lane >> 4;
  const float* pb = w_dn + (size_t)e * D_DIM * H_DIM
                         + (size_t)(d0 + col) * H_DIM + half * 16;

  const f32x8 zero = {0.f,0.f,0.f,0.f,0.f,0.f,0.f,0.f};
  f32x8 acc[4] = {zero, zero, zero, zero};

  const int ldRow = tid >> 2, ldSeg = tid & 3;
  const unsigned long long aSrc =
      (unsigned long long)(uintptr_t)(act + (size_t)sT[ldRow] * H_DIM + ldSeg * 8);
  const unsigned int ldsDst0 = (unsigned int)(uintptr_t)&aT[0][ldRow * LDA + ldSeg * 8];
  const unsigned int ldsDst1 = (unsigned int)(uintptr_t)&aT[1][ldRow * LDA + ldSeg * 8];

  auto mma_tiles = [&](const __bf16* cur, const float* wb_p) {
    bf16x16 b = cvt_b_frag(wb_p);
#pragma unroll
    for (int s = 0; s < 4; ++s) {
      bf16x16 a = lds_a_frag(cur, s, lane);
      acc[s] = __builtin_amdgcn_wmma_f32_16x16x32_bf16(false, a, false, b,
                                                       (short)0, acc[s], false, false);
    }
  };

  async_copy_b128(ldsDst0, aSrc);          // prime buffer 0 (tile 0)
  wait_async0();
  __syncthreads();

#pragma unroll 1
  for (int k0 = 0; k0 < H_DIM - 64; k0 += 64) {
    async_copy_b128(ldsDst1, aSrc + (unsigned long long)(k0 + 32) * 2);
    __builtin_prefetch(pb + k0 + 64, 0, 1);
    mma_tiles(aT[0], pb + k0);
    wait_async0(); __syncthreads();

    async_copy_b128(ldsDst0, aSrc + (unsigned long long)(k0 + 64) * 2);
    __builtin_prefetch(pb + k0 + 96, 0, 1);
    mma_tiles(aT[1], pb + k0 + 32);
    wait_async0(); __syncthreads();
  }
  { // epilogue: last two tiles
    const int k0 = H_DIM - 64;
    async_copy_b128(ldsDst1, aSrc + (unsigned long long)(k0 + 32) * 2);
    mma_tiles(aT[0], pb + k0);
    wait_async0(); __syncthreads();
    mma_tiles(aT[1], pb + k0 + 32);
  }

#pragma unroll
  for (int s = 0; s < 4; ++s) {
#pragma unroll
    for (int r = 0; r < 8; ++r) {
      int m = s * 16 + half * 8 + r;
      if (mBase + m < cn) {
        int t = sT[m] >> 1;   // slot -> token
        __hip_atomic_fetch_add(&out[(size_t)t * D_DIM + d0 + col], acc[s][r],
                               __ATOMIC_RELAXED, __HIP_MEMORY_SCOPE_AGENT);
      }
    }
  }
}

// ---------------------------------------------------------------------------
// Launcher
// ---------------------------------------------------------------------------
extern "C" void kernel_launch(void* const* d_in, const int* in_sizes, int n_in,
                              void* d_out, int out_size, void* d_ws, size_t ws_size,
                              hipStream_t stream) {
  (void)in_sizes; (void)n_in; (void)out_size; (void)ws_size;
  const float* hidden = (const float*)d_in[0];
  const float* tkw    = (const float*)d_in[1];
  const int*   tki    = (const int*)  d_in[2];
  const float* w_up   = (const float*)d_in[3];
  const float* w_dn   = (const float*)d_in[4];
  float* out = (float*)d_out;

  char* ws = (char*)d_ws;
  int*    cnt  = (int*)  (ws + OFF_CNT);
  int*    rows = (int*)  (ws + OFF_ROWS);
  float*  wts  = (float*)(ws + OFF_WTS);
  __bf16* hsb  = (__bf16*)(ws + OFF_HSB);
  __bf16* act  = (__bf16*)(ws + OFF_ACT);

  hipMemsetAsync(cnt, 0, 1024, stream);                                  // expert counters
  hipMemsetAsync(out, 0, (size_t)T_TOK * D_DIM * sizeof(float), stream); // atomic accum target

  moe_route_kernel<<<(T_TOK * K_TOP + 255) / 256, 256, 0, stream>>>(tki, tkw, cnt, rows, wts);
  moe_cvt_kernel  <<<(T_TOK * D_DIM / 4) / 256, 256, 0, stream>>>(hidden, hsb);

  dim3 gUp(MAXR / 64, H_DIM / 128, E_EXP);   // most blocks early-exit on cnt[e]
  moe_up_kernel<<<gUp, 256, 0, stream>>>(hsb, w_up, rows, wts, cnt, act);

  dim3 gDn(MAXR / 64, D_DIM / 128, E_EXP);
  moe_dn_kernel<<<gDn, 256, 0, stream>>>(act, w_dn, rows, cnt, out);
}
